// AutoregressiveFlow_47734266528050
// MI455X (gfx1250) — matrix operand order repaired and checked by reading, not compiled, not run on hardware
//
#include <hip/hip_runtime.h>

typedef __attribute__((ext_vector_type(16))) _Float16 v16h;
typedef __attribute__((ext_vector_type(8)))  float    v8f;

namespace {
constexpr int Tt    = 50;
constexpr int Dd    = 2;
constexpr int Hh    = 64;
constexpr int PADW2 = 104;  // s_whh row stride (f16): rows hold K=0..95 ; 208B = 16B aligned, 52-dword bank stride
constexpr int PADW  = 88;   // s_w1 / s_hA row stride: 176B aligned, 44-dword bank stride
constexpr int PADN  = 40;   // s_win / s_w2 / s_hid row stride: 80B aligned, 20-dword bank stride
constexpr int WAVES = 4;
constexpr int Bb    = 262144;
}

union alignas(32) HVec {
  v16h v;
  uint4 q[2];
};

__device__ __forceinline__ v8f splat8(float s) {
  v8f r;
#pragma unroll
  for (int i = 0; i < 8; ++i) r[i] = s;
  return r;
}

// Build a v16h WMMA operand from two 16B runs of f16 in LDS.
__device__ __forceinline__ v16h load_ab(const _Float16* p0, const _Float16* p1) {
  HVec u;
  u.q[0] = *(const uint4*)p0;
  u.q[1] = *(const uint4*)p1;
  return u.v;
}

__device__ __forceinline__ v8f wmma_f16(v16h a, v16h b, v8f c) {
  return __builtin_amdgcn_wmma_f32_16x16x32_f16(false, a, false, b, (short)0, c, false, false);
}

// Hardware tanh (CDNA5 TRANS op). Prefer builtin; else emit v_tanh_f32 directly.
// Trailing v_nop covers the TRANS result-use co-execution hazard for the asm path.
__device__ __forceinline__ float fast_tanh(float x) {
#if __has_builtin(__builtin_amdgcn_tanhf)
  return __builtin_amdgcn_tanhf(x);
#elif __has_builtin(__builtin_amdgcn_tanh_f32)
  return __builtin_amdgcn_tanh_f32(x);
#else
  float r;
  asm volatile("v_tanh_f32 %0, %1\n\tv_nop" : "=v"(r) : "v"(x));
  return r;
#endif
}

// sigmoid(x) = 0.5*tanh(0.5x) + 0.5  (1 TRANS instead of 2)
__device__ __forceinline__ float fast_sigmoid(float x) {
  return __builtin_fmaf(0.5f, fast_tanh(0.5f * x), 0.5f);
}

__global__ __launch_bounds__(WAVES * 32) void arflow_kernel(
    const float* __restrict__ x, const float* __restrict__ z,
    const float* __restrict__ w_ih, const float* __restrict__ w_hh,
    const float* __restrict__ b_ih, const float* __restrict__ b_hh,
    const float* __restrict__ W1, const float* __restrict__ b1,
    const float* __restrict__ W2, const float* __restrict__ b2,
    float* __restrict__ y_out, float* __restrict__ ld_out)
{
  // B matrices stored row-major by output neuron: s_*[n][k] = B[k][n]
  __shared__ alignas(16) _Float16 s_whh[192 * PADW2];      // k=0..63: w_hh; k=64,65: w_ih (r/z); k=66: bias; else 0
  __shared__ alignas(16) _Float16 s_w1[32 * PADW];         // W1[n][k]
  __shared__ alignas(16) _Float16 s_w2[16 * PADN];         // W2 padded N 4->16, K=32
  __shared__ alignas(16) _Float16 s_win[64 * PADN];        // i_n path: k=0,1: w_ih(n-gate); k=2: b_ih(n); else 0
  __shared__ alignas(16) _Float16 s_hA[WAVES][16 * PADW];  // h staged row-major [M][k] f16
  __shared__ alignas(16) _Float16 s_hid[WAVES][16 * PADN]; // hid staged row-major [M][k] f16
  __shared__ alignas(16) float    s_x[WAVES][16 * Tt * Dd];
  __shared__ alignas(16) float    s_y[WAVES][32];          // y_{t-1}: [M][d]

  const int tid  = threadIdx.x;
  const int wave = tid >> 5;
  const int lane = tid & 31;
  const int c    = lane & 15;
  const int hi   = lane >> 4;
  const int waveRow = blockIdx.x * (WAVES * 16) + wave * 16;

  // ---- stage weights into LDS (once per workgroup) ----
  for (int i = tid; i < 192 * 96; i += blockDim.x) {
    int g = i / 96, k = i - g * 96;
    float v;
    if (k < 64)        v = w_hh[g * 64 + k];
    else if (k == 64)  v = (g < 128) ? w_ih[g * 2 + 0] : 0.0f;
    else if (k == 65)  v = (g < 128) ? w_ih[g * 2 + 1] : 0.0f;
    else if (k == 66)  v = (g < 128) ? (b_ih[g] + b_hh[g]) : 0.0f;
    else               v = 0.0f;
    s_whh[g * PADW2 + k] = (_Float16)v;
  }
  for (int i = tid; i < 32 * 64; i += blockDim.x) {
    int n = i >> 6, k = i & 63;
    s_w1[n * PADW + k] = (_Float16)W1[i];
  }
  for (int i = tid; i < 16 * 32; i += blockDim.x) {
    int n = i >> 5, k = i & 31;
    s_w2[n * PADN + k] = (_Float16)((n < 4) ? W2[n * 32 + k] : 0.0f);
  }
  for (int i = tid; i < 64 * 32; i += blockDim.x) {
    int gn = i >> 5, k = i & 31;
    int g = 128 + gn;
    float v = (k == 0) ? w_ih[g * 2 + 0]
            : (k == 1) ? w_ih[g * 2 + 1]
            : (k == 2) ? b_ih[g] : 0.0f;
    s_win[gn * PADN + k] = (_Float16)v;
  }
  __syncthreads();

  // ---- stage this wave's x slice (16 consecutive rows contiguous: coalesced) ----
  {
    const float* xp = x + (size_t)waveRow * (Tt * Dd);
    float* sxw = s_x[wave];
    for (int i = lane; i < 16 * Tt * Dd; i += 32) sxw[i] = xp[i];
  }
  s_y[wave][lane] = 0.0f;

  // remaining per-lane constants
  float bhn[4];
#pragma unroll
  for (int j = 0; j < 4; ++j) bhn[j] = b_hh[128 + 16 * j + c];
  const float b1c0 = b1[c];
  const float b1c1 = b1[16 + c];
  const float b2c  = (c < 4) ? b2[c] : 0.0f;

  // ---- h init from z (D-layout registers) + seed A-staging LDS ----
  float hprev[4][8];
  _Float16* hA  = s_hA[wave];
  _Float16* hid = s_hid[wave];
  float*    sy  = s_y[wave];
  const float* sx = s_x[wave];
#pragma unroll
  for (int j = 0; j < 4; ++j)
#pragma unroll
    for (int r = 0; r < 8; ++r) {
      float hv = z[(size_t)(waveRow + r + 8 * hi) * Hh + 16 * j + c];
      hprev[j][r] = hv;
      hA[(r + 8 * hi) * PADW + 16 * j + c] = (_Float16)hv;
    }

  // running product of scales (matches reference: log(abs(prod_t scale)))
  float ldprod[8];
#pragma unroll
  for (int r = 0; r < 8; ++r) ldprod[r] = 1.0f;

  const int kb  = 16 * hi;   // A-operand K base
  const int kbB = 16 * hi;   // B-operand K base
  const v8f zc = splat8(0.0f);

#pragma unroll 1
  for (int t = 0; t < Tt; ++t) {
    // A chunk-2 operand: row M = [y_{t-1}[M][0], y_{t-1}[M][1], 1, 0...]
    v16h a2;
#pragma unroll
    for (int e = 0; e < 16; ++e) a2[e] = (_Float16)0.0f;
    if (hi == 0) {
      a2[0] = (_Float16)sy[c * 2 + 0];
      a2[1] = (_Float16)sy[c * 2 + 1];
      a2[2] = (_Float16)1.0f;
    }

    // A operand: previous h, f16 [16 x 64]
    v16h hA0 = load_ab(&hA[c * PADW + kb],      &hA[c * PADW + kb + 8]);
    v16h hA1 = load_ab(&hA[c * PADW + 32 + kb], &hA[c * PADW + 32 + kb + 8]);

#pragma unroll
    for (int j = 0; j < 4; ++j) {
      const _Float16* wr = &s_whh[(16 * j + c) * PADW2];
      const _Float16* wz = wr + 64 * PADW2;
      const _Float16* wn = wr + 128 * PADW2;
      const _Float16* wi = &s_win[(16 * j + c) * PADN];

      // r gate: full pre-activation (h, y, bias) via 3 chunks
      v8f aR = wmma_f16(hA0, load_ab(wr + kbB,      wr + kbB + 8),      zc);
      aR     = wmma_f16(hA1, load_ab(wr + 32 + kbB, wr + 32 + kbB + 8), aR);
      aR     = wmma_f16(a2,  load_ab(wr + 64 + kbB, wr + 64 + kbB + 8), aR);
      // z gate
      v8f aZ = wmma_f16(hA0, load_ab(wz + kbB,      wz + kbB + 8),      zc);
      aZ     = wmma_f16(hA1, load_ab(wz + 32 + kbB, wz + 32 + kbB + 8), aZ);
      aZ     = wmma_f16(a2,  load_ab(wz + 64 + kbB, wz + 64 + kbB + 8), aZ);
      // n gate, h-path: h@w_hh_n + b_hh (bias via C-init)
      v8f aN = wmma_f16(hA0, load_ab(wn + kbB,      wn + kbB + 8),      splat8(bhn[j]));
      aN     = wmma_f16(hA1, load_ab(wn + 32 + kbB, wn + 32 + kbB + 8), aN);
      // n gate, i-path: y@w_ih_n + b_ih via the [y0,y1,1] operand
      v8f aI = wmma_f16(a2, load_ab(wi + kbB, wi + kbB + 8), zc);

      // elementwise GRU (D-layout: lane holds rows r+8*hi, col 16j+c)
#pragma unroll
      for (int r = 0; r < 8; ++r) {
        float rg = fast_sigmoid(aR[r]);
        float ug = fast_sigmoid(aZ[r]);
        float nn = fast_tanh(__builtin_fmaf(rg, aN[r], aI[r]));
        float hnew = __builtin_fmaf(ug, hprev[j][r] - nn, nn);
        hprev[j][r] = hnew;
        hA[(r + 8 * hi) * PADW + 16 * j + c] = (_Float16)hnew;
      }
    }

    // ---- MLP layer 1: hid = relu(h_new @ W1^T + b1), N = 32 ----
    v16h nA0 = load_ab(&hA[c * PADW + kb],      &hA[c * PADW + kb + 8]);
    v16h nA1 = load_ab(&hA[c * PADW + 32 + kb], &hA[c * PADW + 32 + kb + 8]);

    const _Float16* w1a = &s_w1[c * PADW];
    const _Float16* w1b = &s_w1[(16 + c) * PADW];
    v8f t0 = splat8(b1c0);
    t0 = wmma_f16(nA0, load_ab(w1a + kbB,      w1a + kbB + 8),      t0);
    t0 = wmma_f16(nA1, load_ab(w1a + 32 + kbB, w1a + 32 + kbB + 8), t0);
    v8f t1 = splat8(b1c1);
    t1 = wmma_f16(nA0, load_ab(w1b + kbB,      w1b + kbB + 8),      t1);
    t1 = wmma_f16(nA1, load_ab(w1b + 32 + kbB, w1b + 32 + kbB + 8), t1);

#pragma unroll
    for (int r = 0; r < 8; ++r) {
      int M = r + 8 * hi;
      float u0 = t0[r] > 0.0f ? t0[r] : 0.0f;
      float u1 = t1[r] > 0.0f ? t1[r] : 0.0f;
      hid[M * PADN + c]      = (_Float16)u0;
      hid[M * PADN + 16 + c] = (_Float16)u1;
    }

    // ---- MLP layer 2: ds = hid @ W2^T + b2 (K=32, N padded 4->16): one WMMA ----
    v16h mA = load_ab(&hid[c * PADN + kb], &hid[c * PADN + kb + 8]);
    const _Float16* w2p = &s_w2[c * PADN];
    v8f dsv = splat8(b2c);
    dsv = wmma_f16(mA, load_ab(w2p + kbB, w2p + kbB + 8), dsv);

    // ---- scale/dloc, y update, scale-product accumulation ----
    float sc[8];
#pragma unroll
    for (int r = 0; r < 8; ++r) {
      float v = dsv[r];
      float e = __expf(v);
      float sp = (v > 20.0f) ? v : __logf(1.0f + e);
      sc[r] = sp + 0.001f;          // meaningful on lanes c==2,3
      ldprod[r] *= sc[r];           // consumed only from lanes 2,3
    }
    float scOth[8];
#pragma unroll
    for (int r = 0; r < 8; ++r) scOth[r] = __shfl_xor(sc[r], 2, 32);

    if (c < 2) {
#pragma unroll
      for (int r = 0; r < 8; ++r) {
        int M = r + 8 * hi;
        float yprev = sy[M * 2 + c];
        float xv = sx[M * (Tt * Dd) + t * Dd + c];
        float ynew = yprev + dsv[r] + scOth[r] * xv;
        sy[M * 2 + c] = ynew;
        y_out[(size_t)(waveRow + M) * (Tt * Dd) + t * Dd + c] = ynew;
      }
    }
  }

  // ---- logabsdet epilogue: log(abs(prod)) per (row,d), then combine d lanes ----
  float lg[8];
#pragma unroll
  for (int r = 0; r < 8; ++r) lg[r] = __logf(__builtin_fabsf(ldprod[r]));
  float ldo[8];
#pragma unroll
  for (int r = 0; r < 8; ++r) ldo[r] = __shfl_xor(lg[r], 1, 32);
  if (c == 2) {
#pragma unroll
    for (int r = 0; r < 8; ++r)
      ld_out[waveRow + r + 8 * hi] = lg[r] + ldo[r];
  }
}

extern "C" void kernel_launch(void* const* d_in, const int* in_sizes, int n_in,
                              void* d_out, int out_size, void* d_ws, size_t ws_size,
                              hipStream_t stream) {
  (void)in_sizes; (void)n_in; (void)out_size; (void)d_ws; (void)ws_size;
  const float* x    = (const float*)d_in[0];
  const float* z    = (const float*)d_in[1];
  const float* w_ih = (const float*)d_in[2];
  const float* w_hh = (const float*)d_in[3];
  const float* b_ih = (const float*)d_in[4];
  const float* b_hh = (const float*)d_in[5];
  const float* W1   = (const float*)d_in[6];
  const float* b1   = (const float*)d_in[7];
  const float* W2   = (const float*)d_in[8];
  const float* b2   = (const float*)d_in[9];

  float* y_out  = (float*)d_out;
  float* ld_out = y_out + (size_t)Bb * Tt * Dd;

  dim3 grid(Bb / (WAVES * 16));   // 4096 workgroups, 64 batch rows each
  dim3 block(WAVES * 32);         // 4 waves of 32
  arflow_kernel<<<grid, block, 0, stream>>>(x, z, w_ih, w_hh, b_ih, b_hh,
                                            W1, b1, W2, b2, y_out, ld_out);
}
